// SpGAT_84670985273722
// MI455X (gfx1250) — matrix operand order repaired and checked by reading, not compiled, not run on hardware
//
#include <hip/hip_runtime.h>
#include <stdint.h>

// ---------------- problem constants (match reference) ----------------
#define N_NODES 50000
#define NFEAT   512
#define NHID    64
#define NHEADS  8
#define NCLASS  32
#define NEDGE   800000
#define BATCH   20000
#define NNZ     32
#define ALPHA   0.2f

// ---------------- CDNA5 WMMA types ----------------
typedef __attribute__((ext_vector_type(16))) __bf16 v16bf;
typedef __attribute__((ext_vector_type(8)))  float  v8f;

union Frag { uint4 u[2]; v16bf v; };

// ---------------- CDNA5 async global->LDS copy (ASYNCcnt path) ----------------
#if defined(__has_builtin)
#if __has_builtin(__builtin_amdgcn_global_load_async_to_lds_b128) && \
    __has_builtin(__builtin_amdgcn_s_wait_asynccnt)
#define HAVE_ASYNC 1
#endif
#endif
#ifndef HAVE_ASYNC
#define HAVE_ASYNC 0
#endif

#if HAVE_ASYNC
typedef int v4i_vs __attribute__((vector_size(16)));
typedef __attribute__((address_space(1))) v4i_vs* gptr128_t;
typedef __attribute__((address_space(3))) v4i_vs* lptr128_t;
#endif

__device__ __forceinline__ void async_cp16(unsigned short* lds, const unsigned short* g) {
#if HAVE_ASYNC
  // flat->AS casts via integer: flat global == AS1 address; flat LDS low 32 bits == LDS offset
  __builtin_amdgcn_global_load_async_to_lds_b128(
      (gptr128_t)(uintptr_t)g, (lptr128_t)(uintptr_t)lds, 0, 0);
#else
  *(uint4*)lds = *(const uint4*)g;
#endif
}

__device__ __forceinline__ void async_wait_all() {
#if HAVE_ASYNC
  __builtin_amdgcn_s_wait_asynccnt(0);
#endif
}

__device__ __forceinline__ unsigned short f2bf(float f) {
  unsigned u = __float_as_uint(f);
  u += 0x7fffu + ((u >> 16) & 1u);          // round-to-nearest-even
  return (unsigned short)(u >> 16);
}

// ---------------- utility kernels ----------------
__global__ void k_zero(float* __restrict__ p, long n) {
  long i = (long)blockIdx.x * blockDim.x + threadIdx.x;
  long st = (long)gridDim.x * blockDim.x;
  for (; i < n; i += st) p[i] = 0.f;
}

__global__ void k_cvt_bf16(const float4* __restrict__ s, ushort4* __restrict__ d, long n4) {
  long i = (long)blockIdx.x * blockDim.x + threadIdx.x;
  long st = (long)gridDim.x * blockDim.x;
  for (; i < n4; i += st) {
    float4 f = s[i];
    ushort4 o;
    o.x = f2bf(f.x); o.y = f2bf(f.y); o.z = f2bf(f.z); o.w = f2bf(f.w);
    d[i] = o;
  }
}

// W_heads [8][512][64] -> w1t [512 out][512 k] bf16 (out = head*64+col), transposed for B-frag loads
__global__ void k_pack_w1(const float* __restrict__ W, unsigned short* __restrict__ Wt) {
  int idx = blockIdx.x * blockDim.x + threadIdx.x;
  if (idx >= 512 * 512) return;
  int n = idx >> 9, k = idx & 511;
  int head = n >> 6, col = n & 63;
  Wt[(long)n * 512 + k] = f2bf(W[((long)head * NFEAT + k) * NHID + col]);
}

// W_out [512][64] -> w2t [64 out][512 k] bf16
__global__ void k_pack_w2(const float* __restrict__ W, unsigned short* __restrict__ Wt) {
  int idx = blockIdx.x * blockDim.x + threadIdx.x;
  if (idx >= 64 * 512) return;
  int n = idx >> 9, k = idx & 511;
  Wt[(long)n * 512 + k] = f2bf(W[(long)k * NHID + n]);
}

// ---------------- bf16 WMMA GEMM: C[M,Ncols] = A[M,512] * Bt[Ncols,512]^T ----------------
// Block: 256 threads = 8 waves. Block tile M=128, N=64. Wave w owns rows w*16..w*16+15,
// all 64 cols (4 accumulators of 16x16). K loop in steps of 32 (one WMMA-K per step).
// Double-buffered LDS with async global->LDS copies: DMA of tile t+1 overlaps WMMA on tile t,
// and only ONE barrier per K-step is needed.
__launch_bounds__(256)
__global__ void k_gemm_bf16(const unsigned short* __restrict__ A,
                            const unsigned short* __restrict__ Bt,
                            float* __restrict__ C, int M, int Ncols) {
  __shared__ unsigned short At[2 * 128 * 32];   // 2 x 8 KB, row-major [m][k]
  __shared__ unsigned short Bts[2 * 64 * 32];   // 2 x 4 KB, N-major  [n][k]

  const int tid  = threadIdx.x;
  const int wid  = tid >> 5;
  const int lane = tid & 31;
  const int row0 = blockIdx.x * 128;
  const int n0   = blockIdx.y * 64;

  v8f c[4] = {};

  const int arow = tid >> 1;            // 0..127
  const int acol = (tid & 1) * 16;      // ushort units
  const int brow = tid >> 2;            // 0..63
  const int bcol = (tid & 3) * 8;       // ushort units
  const bool arow_ok = (row0 + arow) < M;
  const long gabase = (long)(row0 + arow) * 512 + acol;
  const long gbbase = (long)(n0 + brow) * 512 + bcol;

  // zero-fill out-of-range A rows once (both buffers); asyncs never touch them
  if (!arow_ok) {
    uint4 z = {0u, 0u, 0u, 0u};
#pragma unroll
    for (int p = 0; p < 2; ++p) {
      *(uint4*)(At + p * 4096 + arow * 32 + acol)     = z;
      *(uint4*)(At + p * 4096 + arow * 32 + acol + 8) = z;
    }
  }

  // prologue: stage K-tile 0 into buffer 0
  if (arow_ok) {
    async_cp16(At + arow * 32 + acol,     A + gabase);
    async_cp16(At + arow * 32 + acol + 8, A + gabase + 8);
  }
  async_cp16(Bts + brow * 32 + bcol, Bt + gbbase);

  for (int t = 0; t < 16; ++t) {
    async_wait_all();        // own async copies for tile t complete
    __syncthreads();         // all waves' tile-t data visible; tile t-1 reads done

    const int buf  = t & 1;
    const int nbuf = buf ^ 1;

    // kick off DMA for tile t+1 into the other buffer (overlaps the WMMAs below)
    if (t + 1 < 16) {
      const long ka = gabase + (t + 1) * 32;
      const long kb = gbbase + (t + 1) * 32;
      if (arow_ok) {
        async_cp16(At + nbuf * 4096 + arow * 32 + acol,     A + ka);
        async_cp16(At + nbuf * 4096 + arow * 32 + acol + 8, A + ka + 8);
      }
      async_cp16(Bts + nbuf * 2048 + brow * 32 + bcol, Bt + kb);
      // prefetch tile t+2 toward L2 (global_prefetch_b8)
      if (t + 2 < 16) {
        if (arow_ok) __builtin_prefetch(A + gabase + (t + 2) * 32, 0, 1);
        __builtin_prefetch(Bt + gbbase + (t + 2) * 32, 0, 1);
      }
    }

    // ---- fragments per ISA 7.12.2 16-bit layouts ----
    const unsigned short* Ab = At + buf * 4096;
    const unsigned short* Bb = Bts + buf * 2048;
    const int mrow = wid * 16 + (lane & 15);
    const int half = lane >> 4;
    Frag a;
    a.u[0] = *(const uint4*)(Ab + mrow * 32 + half * 8);        // K 0..7 / 8..15
    a.u[1] = *(const uint4*)(Ab + mrow * 32 + 16 + half * 8);   // K 16..23 / 24..31
#pragma unroll
    for (int j = 0; j < 4; ++j) {
      const int nl = j * 16 + (lane & 15);
      Frag b;
      b.u[0] = *(const uint4*)(Bb + nl * 32 + half * 8);
      b.u[1] = *(const uint4*)(Bb + nl * 32 + 16 + half * 8);
      c[j] = __builtin_amdgcn_wmma_f32_16x16x32_bf16(
          false, a.v, false, b.v, (short)0, c[j], false, false);
    }
  }

  // ---- store C (8 VGPRs: M = (lane>=16 ? 8:0)+r, N = lane&15) ----
  const int mbase = row0 + wid * 16 + (lane >> 4) * 8;
  const int col   = lane & 15;
#pragma unroll
  for (int j = 0; j < 4; ++j) {
#pragma unroll
    for (int r = 0; r < 8; ++r) {
      int gr = mbase + r;
      if (gr < M) C[(long)gr * Ncols + n0 + j * 16 + col] = c[j][r];
    }
  }
}

// ---------------- attention scores: s_src=h.a[:64], s_dst=h.a[64:128] ----------------
__global__ void k_scores(const float* __restrict__ hmat, int rowstride,
                         const float* __restrict__ avec,
                         float* __restrict__ ssrc, float* __restrict__ sdst,
                         int nheads, int nnodes) {
  int wg = blockIdx.x * 8 + (threadIdx.x >> 5);
  if (wg >= nnodes * nheads) return;
  int lane = threadIdx.x & 31;
  int i = wg / nheads;
  int h = wg - i * nheads;
  const float* hp = hmat + (long)i * rowstride + h * 64;
  const float* ap = avec + h * 128;
  float s0 = hp[lane] * ap[lane]      + hp[lane + 32] * ap[lane + 32];
  float s1 = hp[lane] * ap[64 + lane] + hp[lane + 32] * ap[96 + lane];
  for (int off = 16; off; off >>= 1) {
    s0 += __shfl_xor(s0, off, 32);
    s1 += __shfl_xor(s1, off, 32);
  }
  if (lane == 0) {
    ssrc[(long)h * nnodes + i] = s0;
    sdst[(long)h * nnodes + i] = s1;
  }
}

// ---------------- layer-1 edge aggregation: wave per edge, all 8 heads ----------------
__global__ void k_edge1(const int* __restrict__ ei, const float* __restrict__ h1,
                        const float* __restrict__ ssrc, const float* __restrict__ sdst,
                        float* __restrict__ acc, float* __restrict__ den) {
  int e = blockIdx.x * 8 + (threadIdx.x >> 5);
  if (e >= NEDGE) return;
  int lane = threadIdx.x & 31;
  int s = ei[e], d = ei[NEDGE + e];
  const float* hd = h1 + (long)d * 512;
  float* ac = acc + (long)s * 512;
#pragma unroll
  for (int h = 0; h < NHEADS; ++h) {
    float sc = ssrc[(long)h * N_NODES + s] + sdst[(long)h * N_NODES + d];
    float lr = sc > 0.f ? sc : ALPHA * sc;
    float w  = __expf(-lr);
    if (lane == 0) atomicAdd(&den[(long)h * N_NODES + s], w);
    atomicAdd(&ac[h * 64 + lane],      w * hd[h * 64 + lane]);
    atomicAdd(&ac[h * 64 + lane + 32], w * hd[h * 64 + lane + 32]);
  }
}

// normalize + per-head ELU + convert to bf16 (input of layer-2 GEMM)
__global__ void k_norm1(const float* __restrict__ acc, const float* __restrict__ den,
                        unsigned short* __restrict__ x1b) {
  long n = (long)N_NODES * 512;
  long idx = (long)blockIdx.x * blockDim.x + threadIdx.x;
  long st = (long)gridDim.x * blockDim.x;
  for (; idx < n; idx += st) {
    long i = idx >> 9;
    int  c = (int)(idx & 511);
    int  h = c >> 6;
    float v = acc[idx] / (den[(long)h * N_NODES + i] + 1e-16f);
    v = v > 0.f ? v : __expf(v) - 1.f;
    x1b[idx] = f2bf(v);
  }
}

// ---------------- layer-2 edge aggregation (single head, D=64) ----------------
__global__ void k_edge2(const int* __restrict__ ei, const float* __restrict__ h2,
                        const float* __restrict__ ssrc, const float* __restrict__ sdst,
                        float* __restrict__ acc, float* __restrict__ den) {
  int e = blockIdx.x * 8 + (threadIdx.x >> 5);
  if (e >= NEDGE) return;
  int lane = threadIdx.x & 31;
  int s = ei[e], d = ei[NEDGE + e];
  float sc = ssrc[s] + sdst[d];
  float lr = sc > 0.f ? sc : ALPHA * sc;
  float w  = __expf(-lr);
  if (lane == 0) atomicAdd(&den[s], w);
  atomicAdd(&acc[(long)s * 64 + lane],      w * h2[(long)d * 64 + lane]);
  atomicAdd(&acc[(long)s * 64 + lane + 32], w * h2[(long)d * 64 + lane + 32]);
}

__global__ void k_gatfin(const float* __restrict__ acc, const float* __restrict__ den,
                         float* __restrict__ gat) {
  long n = (long)N_NODES * 64;
  long idx = (long)blockIdx.x * blockDim.x + threadIdx.x;
  long st = (long)gridDim.x * blockDim.x;
  for (; idx < n; idx += st) {
    long i = idx >> 6;
    float v = acc[idx] / (den[i] + 1e-16f);
    gat[idx] = v > 0.f ? v : __expf(v) - 1.f;   // final elu
  }
}

// ---------------- FM second-order interaction ----------------
__global__ void k_fm(const int* __restrict__ nzi, const float* __restrict__ nzv,
                     const float* __restrict__ fmV, float* __restrict__ fm) {
  int b = blockIdx.x;
  int d = threadIdx.x;             // 64 threads
  float ss = 0.f, sq = 0.f;
  for (int f = 0; f < NNZ; ++f) {
    int   idx = nzi[b * NNZ + f];
    float val = nzv[b * NNZ + f];
    float v = fmV[(long)idx * NHID + d] * val;
    ss += v; sq += v * v;
  }
  fm[(long)b * NHID + d] = 0.5f * (ss * ss - sq);
}

// ---------------- AFM fusion + classifier + log_softmax (wave per row) ----------------
__global__ void k_afm(const int* __restrict__ tidx, const float* __restrict__ gat,
                      const float* __restrict__ fm,
                      const float* __restrict__ attn_W, const float* __restrict__ attn_b,
                      const float* __restrict__ proj_W, const float* __restrict__ proj_b,
                      const float* __restrict__ final_W, const float* __restrict__ final_b,
                      float* __restrict__ out) {
  int b = blockIdx.x * 8 + (threadIdx.x >> 5);
  if (b >= BATCH) return;
  int lane = threadIdx.x & 31;
  const float* gr = gat + (long)tidx[b] * NHID;
  const float* fr = fm  + (long)b * NHID;

  float sc[2];
#pragma unroll
  for (int t = 0; t < 2; ++t) {
    const float* field = t ? fr : gr;
    float a0 = attn_b[lane], a1 = attn_b[lane + 32];
    for (int k = 0; k < 64; ++k) {
      float fv = field[k];
      a0 += fv * attn_W[k * 64 + lane];
      a1 += fv * attn_W[k * 64 + lane + 32];
    }
    float p = fmaxf(a0, 0.f) * proj_W[lane] + fmaxf(a1, 0.f) * proj_W[lane + 32];
    for (int off = 16; off; off >>= 1) p += __shfl_xor(p, off, 32);
    sc[t] = p + proj_b[0];
  }
  float m  = fmaxf(sc[0], sc[1]);
  float e0 = __expf(sc[0] - m), e1 = __expf(sc[1] - m);
  float w0 = e0 / (e0 + e1), w1 = e1 / (e0 + e1);

  float lg = final_b[lane];                       // NCLASS == 32 == wave32
  for (int k = 0; k < 64; ++k) lg += (w0 * gr[k]) * final_W[k * NCLASS + lane];
  for (int k = 0; k < 64; ++k) lg += (w1 * fr[k]) * final_W[(64 + k) * NCLASS + lane];

  float mx = lg;
  for (int off = 16; off; off >>= 1) mx = fmaxf(mx, __shfl_xor(mx, off, 32));
  float ex = __expf(lg - mx), se = ex;
  for (int off = 16; off; off >>= 1) se += __shfl_xor(se, off, 32);
  out[(long)b * NCLASS + lane] = lg - mx - __logf(se);
}

// ---------------- host orchestration ----------------
extern "C" void kernel_launch(void* const* d_in, const int* in_sizes, int n_in,
                              void* d_out, int out_size, void* d_ws, size_t ws_size,
                              hipStream_t stream) {
  (void)in_sizes; (void)n_in; (void)out_size; (void)ws_size;
  const float* x       = (const float*)d_in[0];
  const int*   ei      = (const int*)  d_in[1];
  const int*   tidx    = (const int*)  d_in[2];
  const int*   nzi     = (const int*)  d_in[3];
  const float* nzv     = (const float*)d_in[4];
  const float* W_heads = (const float*)d_in[5];
  const float* a_heads = (const float*)d_in[6];
  const float* W_out   = (const float*)d_in[7];
  const float* a_out   = (const float*)d_in[8];
  const float* fmV     = (const float*)d_in[9];
  const float* attn_W  = (const float*)d_in[10];
  const float* attn_b  = (const float*)d_in[11];
  const float* proj_W  = (const float*)d_in[12];
  const float* proj_b  = (const float*)d_in[13];
  const float* final_W = (const float*)d_in[14];
  const float* final_b = (const float*)d_in[15];
  float* out = (float*)d_out;

  char* ws = (char*)d_ws;
  size_t off = 0;
  auto alloc = [&](size_t bytes) -> char* {
    char* p = ws + off;
    off += (bytes + 255) & ~(size_t)255;
    return p;
  };
  unsigned short* xb   = (unsigned short*)alloc((size_t)N_NODES * 512 * 2); // reused as x1 bf16
  unsigned short* w1t  = (unsigned short*)alloc((size_t)512 * 512 * 2);
  float* h1   = (float*)alloc((size_t)N_NODES * 512 * 4);   // reused as h2
  float* acc1 = (float*)alloc((size_t)N_NODES * 512 * 4);   // reused as acc2
  float* s1s  = (float*)alloc((size_t)NHEADS * N_NODES * 4);
  float* s1d  = (float*)alloc((size_t)NHEADS * N_NODES * 4);
  float* den1 = (float*)alloc((size_t)NHEADS * N_NODES * 4);
  unsigned short* w2t = (unsigned short*)alloc((size_t)64 * 512 * 2);
  float* s2s  = (float*)alloc((size_t)N_NODES * 4);
  float* s2d  = (float*)alloc((size_t)N_NODES * 4);
  float* den2 = (float*)alloc((size_t)N_NODES * 4);
  float* gat  = (float*)alloc((size_t)N_NODES * 64 * 4);
  float* fmf  = (float*)alloc((size_t)BATCH * 64 * 4);

  // ---- stage bf16 operands ----
  k_cvt_bf16<<<4096, 256, 0, stream>>>((const float4*)x, (ushort4*)xb,
                                       (long)N_NODES * 512 / 4);
  k_pack_w1<<<(512 * 512 + 255) / 256, 256, 0, stream>>>(W_heads, w1t);

  // ---- layer 1: fused 8-head GEMM [50000,512]@[512,512] ----
  dim3 g1((N_NODES + 127) / 128, 8);
  k_gemm_bf16<<<g1, 256, 0, stream>>>(xb, w1t, h1, N_NODES, 512);
  k_scores<<<(N_NODES * NHEADS + 7) / 8, 256, 0, stream>>>(h1, 512, a_heads,
                                                           s1s, s1d, NHEADS, N_NODES);
  k_zero<<<8192, 256, 0, stream>>>(acc1, (long)N_NODES * 512);
  k_zero<<<2048, 256, 0, stream>>>(den1, (long)NHEADS * N_NODES);
  k_edge1<<<(NEDGE + 7) / 8, 256, 0, stream>>>(ei, h1, s1s, s1d, acc1, den1);
  k_norm1<<<8192, 256, 0, stream>>>(acc1, den1, xb /* x1 bf16 */);

  // ---- layer 2 GEMM [50000,512]@[512,64] ----
  k_pack_w2<<<(64 * 512 + 255) / 256, 256, 0, stream>>>(W_out, w2t);
  dim3 g2((N_NODES + 127) / 128, 1);
  k_gemm_bf16<<<g2, 256, 0, stream>>>(xb, w2t, h1 /* h2 */, N_NODES, 64);
  k_scores<<<(N_NODES + 7) / 8, 256, 0, stream>>>(h1, 64, a_out, s2s, s2d, 1, N_NODES);
  k_zero<<<2048, 256, 0, stream>>>(acc1, (long)N_NODES * 64);
  k_zero<<<256, 256, 0, stream>>>(den2, (long)N_NODES);
  k_edge2<<<(NEDGE + 7) / 8, 256, 0, stream>>>(ei, h1, s2s, s2d, acc1, den2);
  k_gatfin<<<2048, 256, 0, stream>>>(acc1, den2, gat);

  // ---- FM + AFM fusion + classifier ----
  k_fm<<<BATCH, 64, 0, stream>>>(nzi, nzv, fmV, fmf);
  k_afm<<<(BATCH + 7) / 8, 256, 0, stream>>>(tidx, gat, fmf, attn_W, attn_b,
                                             proj_W, proj_b, final_W, final_b, out);
}
